// Decoder_DIBS_27436251086911
// MI455X (gfx1250) — matrix operand order repaired and checked by reading, not compiled, not run on hardware
//
#include <hip/hip_runtime.h>
#include <math.h>

// ---------------- problem constants ----------------
#define D_N    64
#define K_N    64
#define P_N    32
#define S_N    16
#define NOBS   2048
#define PROJ_N 128
#define ALPHA_C 0.2f
#define H_C     5.0f
#define STEP_C  0.005f

typedef __attribute__((ext_vector_type(16))) _Float16 v16h;
typedef __attribute__((ext_vector_type(8)))  float    v8f;
typedef int v4i __attribute__((vector_size(16)));

#define AS1 __attribute__((address_space(1)))
#define AS3 __attribute__((address_space(3)))

// Async global->LDS path (gfx1250). Probe round 2 confirmed the builtin exists
// with signature (global int4* src, local int4* dst, imm offset, imm cpol).
#if defined(__gfx1250__) && __has_builtin(__builtin_amdgcn_global_load_async_to_lds_b128) && __has_builtin(__builtin_amdgcn_s_wait_asynccnt)
#define USE_ASYNC_LDS 1
#else
#define USE_ASYNC_LDS 0
#endif

// ---------------- RNG (counter-based, deterministic) ----------------
__device__ __forceinline__ unsigned pcg_hash(unsigned x) {
  x = x * 747796405u + 2891336453u;
  unsigned w = ((x >> ((x >> 28u) + 4u)) ^ x) * 277803737u;
  return (w >> 22u) ^ w;
}
__device__ __forceinline__ float u01(unsigned h) {
  return (float)(h >> 8) * (1.0f / 16777216.0f);
}

// ---------------- WMMA 16x16 tile helpers (f16 in, f32 acc) ----------------
// Fragment layouts per CDNA5 ISA 7.12.2 (wave32):
//  A 16x32 f16 : lane m=l&15 ; elem e -> k = (e>>3)*16 + (l>=16)*8 + (e&7)
//  B 32x16 f16 : lane n=l&15 ; elem e -> k = (e>>3)*16 + (l>=16)*8 + (e&7)
//  C/D 16x16 f32: lane n=l&15 ; vgpr r -> m = (l>=16)*8 + r
__device__ __forceinline__ v8f wmma_tile(const float* __restrict__ A, int ars, int acs,
                                         const float* __restrict__ B, int brs, int bcs,
                                         int m0, int n0, int K, v8f acc) {
  const int lane = threadIdx.x & 31;
  const int half = lane >> 4;
  const int mr = m0 + (lane & 15);
  const int nc = n0 + (lane & 15);
  for (int k0 = 0; k0 < K; k0 += 32) {
    v16h a, b;
#pragma unroll
    for (int e = 0; e < 16; ++e) {
      int ka = k0 + ((e >> 3) << 4) + (half << 3) + (e & 7);
      int kb = k0 + ((e >> 3) << 4) + (half << 3) + (e & 7);
      float av = (ka < K) ? A[mr * ars + ka * acs] : 0.0f;
      float bv = (kb < K) ? B[kb * brs + nc * bcs] : 0.0f;
      a[e] = (_Float16)av;
      b[e] = (_Float16)bv;
    }
    acc = __builtin_amdgcn_wmma_f32_16x16x32_f16(false, a, false, b,
                                                 (short)0, acc, false, false);
  }
  return acc;
}

__device__ __forceinline__ void wmma_store(float* __restrict__ C, int crs, int ccs,
                                           int m0, int n0, v8f acc) {
  const int lane = threadIdx.x & 31;
  const int nc = n0 + (lane & 15);
  const int mb = (lane >> 4) << 3;
#pragma unroll
  for (int r = 0; r < 8; ++r)
    C[(m0 + mb + r) * crs + nc * ccs] = acc[r];
}

// ---------------- kernels ----------------

// xbar[j] = mean over observations
__global__ void k_xbar(const float* __restrict__ data, float* __restrict__ xbar) {
  int j = threadIdx.x;
  float s = 0.f;
  for (int i = 0; i < NOBS; ++i) s += data[i * D_N + j];
  xbar[j] = s * (1.0f / (float)NOBS);
}

// S = X^T X (64x64, K=2048). One workgroup, 16 waves = 16 output tiles.
// K staged in 128-row chunks through LDS via async global->LDS copies; each
// chunk is read once from HBM instead of 16x.
#define GRAM_CHUNK 128
__global__ void __launch_bounds__(512)
k_gram_data(const float* __restrict__ data, float* __restrict__ S) {
  __shared__ float sh[GRAM_CHUNK * D_N];  // 32 KB
  const int tid = threadIdx.x;
  const int wave = tid >> 5;
  const int m0 = (wave >> 2) << 4, n0 = (wave & 3) << 4;
  const int lane = tid & 31, half = lane >> 4;
  const int mr = m0 + (lane & 15), nc = n0 + (lane & 15);
  v8f acc = {};
  for (int k0 = 0; k0 < NOBS; k0 += GRAM_CHUNK) {
    for (int idx = tid; idx < GRAM_CHUNK * D_N / 4; idx += 512) {
      const float* src = data + k0 * D_N + idx * 4;
#if USE_ASYNC_LDS
      __builtin_amdgcn_global_load_async_to_lds_b128(
          (AS1 v4i*)(float*)src, (AS3 v4i*)(sh + idx * 4), 0, 0);
#else
      ((float4*)sh)[idx] = *(const float4*)src;
#endif
    }
#if USE_ASYNC_LDS
    __builtin_amdgcn_s_wait_asynccnt(0);
#endif
    __syncthreads();
    for (int kk = 0; kk < GRAM_CHUNK; kk += 32) {
      v16h a, b;
#pragma unroll
      for (int e = 0; e < 16; ++e) {
        int ka = kk + ((e >> 3) << 4) + (half << 3) + (e & 7);
        int kb = kk + ((e >> 3) << 4) + (half << 3) + (e & 7);
        a[e] = (_Float16)sh[ka * D_N + mr];  // A[m][k] = X[k][m]
        b[e] = (_Float16)sh[kb * D_N + nc];  // B[k][n] = X[k][n]
      }
      acc = __builtin_amdgcn_wmma_f32_16x16x32_f16(false, a, false, b,
                                                   (short)0, acc, false, false);
    }
    __syncthreads();
  }
  wmma_store(S, D_N, 1, m0, n0, acc);
}

// R = small_t*I + S + (-N + N/(N+1)) * outer(xbar,xbar)   (mu0 == 0 in both passes)
__global__ void k_finalize_R(float* __restrict__ R, const float* __restrict__ xbar) {
  int idx = blockIdx.x * blockDim.x + threadIdx.x;
  if (idx >= D_N * D_N) return;
  int i = idx >> 6, j = idx & 63;
  const float small_t = 0.5f;   // 1*(66-64-1)/2
  const float cmu = -(float)NOBS + (float)NOBS / ((float)NOBS + 1.0f);
  float r = R[idx] + cmu * xbar[i] * xbar[j];
  if (i == j) r += small_t;
  R[idx] = r;
}

// scores[p] = u_p @ v_p^T ; u=z[...,0], v=z[...,1], layout [P,D,K,2]
__global__ void k_scores(const float* __restrict__ z, float* __restrict__ scores) {
  int b = blockIdx.x;
  int p = b >> 4, t = b & 15;
  int m0 = (t >> 2) << 4, n0 = (t & 3) << 4;
  const float* zp = z + p * 8192;
  v8f acc = {};
  // A = u: u[i][k] = zp[i*128 + k*2] ; B[k][j] = v[j][k] = (zp+1)[j*128 + k*2]
  acc = wmma_tile(zp, 128, 2, zp + 1, 2, 128, m0, n0, K_N, acc);
  wmma_store(scores + p * 4096, D_N, 1, m0, n0, acc);
}

__global__ void k_probs(const float* __restrict__ scores, float* __restrict__ probs) {
  int idx = blockIdx.x * blockDim.x + threadIdx.x;
  if (idx >= P_N * 4096) return;
  int ij = idx & 4095, i = ij >> 6, j = ij & 63;
  probs[idx] = (i == j) ? 0.f : 1.f / (1.f + expf(-ALPHA_C * scores[idx]));
}

// Bernoulli sample regenerated on demand (never materialized)
__device__ __forceinline__ float g_soft(int p, int s, int i, int j,
                                        const float* __restrict__ probs, unsigned seed) {
  if (i == j) return 0.f;
  unsigned idx = (unsigned)((((p * S_N + s) * D_N + i) * D_N) + j);
  unsigned h = pcg_hash(idx ^ (seed * 0x9E3779B9u + 0x6A09E667u));
  return (u01(h) < probs[(p * D_N + i) * D_N + j]) ? 1.f : 0.f;
}

// BGe log-marginal per graph: one block per graph, cooperative Cholesky of the
// gathered (l+1)x(l+1) principal submatrix (parents first, node j last) gives
// BOTH masked slogdets at once (leading-minor property).
__global__ void __launch_bounds__(64)
k_bge(const float* __restrict__ R, const float* __restrict__ g_direct,
      const float* __restrict__ probs, const int* __restrict__ seedp,
      float* __restrict__ out_logp) {
  __shared__ float Ms[64 * 65];
  __shared__ int   idxs[65];
  __shared__ int   ln_s;
  __shared__ float blksum;
  const int tid = threadIdx.x;
  const int graph = blockIdx.x;
  const unsigned seed = (unsigned)seedp[0];
  const int p = graph >> 4, s = graph & 15;

  if (tid == 0) blksum = 0.f;
  __syncthreads();

  for (int j = 0; j < D_N; ++j) {
    if (tid == 0) {
      int l = 0;
      for (int i = 0; i < D_N; ++i) {
        float gv = g_direct ? g_direct[graph * 4096 + i * D_N + j]
                            : g_soft(p, s, i, j, probs, seed);
        if (i != j && gv > 0.5f) idxs[l++] = i;
      }
      idxs[l] = j;
      ln_s = l;
    }
    __syncthreads();
    const int l = ln_s, n = l + 1;
    for (int t = tid; t < n * n; t += 64) {
      int a = t / n, b = t - a * n;
      Ms[a * 65 + b] = R[idxs[a] * D_N + idxs[b]];
    }
    __syncthreads();
    // right-looking Cholesky, thread c owns column c
    for (int k = 0; k < n; ++k) {
      if (tid == 0) Ms[k * 65 + k] = sqrtf(fmaxf(Ms[k * 65 + k], 1e-30f));
      __syncthreads();
      float dk = Ms[k * 65 + k];
      for (int i = k + 1 + tid; i < n; i += 64) Ms[i * 65 + k] /= dk;
      __syncthreads();
      if (tid > k && tid < n) {
        float lck = Ms[tid * 65 + k];
        for (int i = tid; i < n; ++i) Ms[i * 65 + tid] -= Ms[i * 65 + k] * lck;
      }
      __syncthreads();
    }
    if (tid == 0) {
      float ldp = 0.f;
      for (int i = 0; i < l; ++i) ldp += logf(Ms[i * 65 + i]);
      ldp *= 2.f;
      float ldpj = ldp + 2.f * logf(Ms[l * 65 + l]);
      float lf = (float)l;
      const float LOG_PI = 1.1447298858494002f;
      const float LOG_HALF = -0.6931471805599453f;
      // n+alpha_lambd-d = 2050, alpha_lambd-d = 2, alpha_mu = 1, small_t = 0.5
      float log_gamma = 0.5f * (0.f - logf((float)NOBS + 1.f))
                      + lgammaf(0.5f * (2051.f + lf))
                      - lgammaf(0.5f * (3.f + lf))
                      - 0.5f * (float)NOBS * LOG_PI
                      + 0.5f * (3.f + 2.f * lf) * LOG_HALF;
      blksum += log_gamma + 0.5f * (2050.f + lf) * ldp - 0.5f * (2051.f + lf) * ldpj;
    }
    __syncthreads();
  }
  if (tid == 0) out_logp[graph] = blksum;
}

__global__ void k_softmax(const float* __restrict__ logp, float* __restrict__ w) {
  __shared__ float buf[16];
  int p = blockIdx.x, t = threadIdx.x;
  if (t < 16) buf[t] = logp[p * 16 + t];
  __syncthreads();
  if (t == 0) {
    float m = buf[0];
    for (int i = 1; i < 16; ++i) m = fmaxf(m, buf[i]);
    float sum = 0.f;
    for (int i = 0; i < 16; ++i) { buf[i] = expf(buf[i] - m); sum += buf[i]; }
    float inv = 1.f / sum;
    for (int i = 0; i < 16; ++i) w[p * 16 + i] = buf[i] * inv;
  }
}

// Btot = ALPHA * ( sum_s w_s (g_s - probs) + probs(1-probs)*logit_pe ) off-diag
__global__ void k_btot(const float* __restrict__ probs, const float* __restrict__ w,
                       const int* __restrict__ seedp, float* __restrict__ Btot) {
  int idx = blockIdx.x * blockDim.x + threadIdx.x;
  if (idx >= P_N * 4096) return;
  int p = idx >> 12, ij = idx & 4095, i = ij >> 6, j = ij & 63;
  if (i == j) { Btot[idx] = 0.f; return; }
  unsigned seed = (unsigned)seedp[0];
  float pr = probs[idx];
  float acc = 0.f;
  for (int s = 0; s < S_N; ++s) acc += w[p * 16 + s] * (g_soft(p, s, i, j, probs, seed) - pr);
  const float PE = 4.0f / 63.0f;
  const float logit_pe = logf(PE) - log1pf(-PE);
  Btot[idx] = ALPHA_C * (acc + pr * (1.f - pr) * logit_pe);
}

// grad_u = Btot @ v - K*u ; grad_v = Btot^T @ u - K*v  (1/LATENT_STD^2 == K == 64)
__global__ void k_grads(const float* __restrict__ z, const float* __restrict__ Btot,
                        float* __restrict__ grads) {
  int b = blockIdx.x;
  int p = b >> 5, t = b & 31;
  int is_v = t >> 4; t &= 15;
  int m0 = (t >> 2) << 4, n0 = (t & 3) << 4;
  const float* zp = z + p * 8192;
  const float* B = Btot + p * 4096;
  v8f acc = {};
  if (!is_v) acc = wmma_tile(B, 64, 1, zp + 1, 128, 2, m0, n0, D_N, acc);
  else       acc = wmma_tile(B, 1, 64, zp,     128, 2, m0, n0, D_N, acc);
  int lane = threadIdx.x & 31;
  int nc = n0 + (lane & 15), mb = (lane >> 4) << 3;
#pragma unroll
  for (int r = 0; r < 8; ++r) {
    int mr = m0 + mb + r;
    int off = (mr * 64 + nc) * 2 + is_v;
    grads[p * 8192 + off] = acc[r] - 64.f * zp[off];
  }
}

// G = zf zf^T (32x32, K=8192). One workgroup, 4 waves = 4 tiles; zf chunks
// staged through LDS (async path when available).
__global__ void __launch_bounds__(128)
k_gram_z(const float* __restrict__ z, float* __restrict__ G) {
  __shared__ float sh[P_N * 128];  // 16 KB: 32 particles x 128 K-chunk
  const int tid = threadIdx.x;
  const int wave = tid >> 5;
  const int m0 = (wave >> 1) << 4, n0 = (wave & 1) << 4;
  const int lane = tid & 31, half = lane >> 4;
  const int mr = m0 + (lane & 15), nc = n0 + (lane & 15);
  v8f acc = {};
  for (int k0 = 0; k0 < 8192; k0 += 128) {
    for (int idx = tid; idx < P_N * 32; idx += 128) {  // 1024 float4s
      int row = idx >> 5, c4 = idx & 31;
      const float* src = z + row * 8192 + k0 + c4 * 4;
#if USE_ASYNC_LDS
      __builtin_amdgcn_global_load_async_to_lds_b128(
          (AS1 v4i*)(float*)src, (AS3 v4i*)(sh + row * 128 + c4 * 4), 0, 0);
#else
      ((float4*)(sh + row * 128))[c4] = *(const float4*)src;
#endif
    }
#if USE_ASYNC_LDS
    __builtin_amdgcn_s_wait_asynccnt(0);
#endif
    __syncthreads();
    for (int kk = 0; kk < 128; kk += 32) {
      v16h a, b;
#pragma unroll
      for (int e = 0; e < 16; ++e) {
        int ka = kk + ((e >> 3) << 4) + (half << 3) + (e & 7);
        int kb = kk + ((e >> 3) << 4) + (half << 3) + (e & 7);
        a[e] = (_Float16)sh[mr * 128 + ka];
        b[e] = (_Float16)sh[nc * 128 + kb];
      }
      acc = __builtin_amdgcn_wmma_f32_16x16x32_f16(false, a, false, b,
                                                   (short)0, acc, false, false);
    }
    __syncthreads();
  }
  wmma_store(G, P_N, 1, m0, n0, acc);
}

__global__ void k_kmat(const float* __restrict__ G, float* __restrict__ Km) {
  int idx = blockIdx.x * blockDim.x + threadIdx.x;
  if (idx >= P_N * P_N) return;
  int p = idx >> 5, q = idx & 31;
  float sq = G[p * 32 + p] + G[q * 32 + q] - 2.f * G[p * 32 + q];
  Km[idx] = expf(-sq * (1.0f / H_C));
}

// z_new = z + STEP/P * ( K@(gf - (2/H) zf) + (2/H) zf * rowsum(K) )
__global__ void k_znew(const float* __restrict__ z, const float* __restrict__ grads,
                       const float* __restrict__ Km, float* __restrict__ z_new) {
  int idx = blockIdx.x * blockDim.x + threadIdx.x;
  if (idx >= P_N * 8192) return;
  int p = idx >> 13, e = idx & 8191;
  const float th = 2.0f / H_C;
  float acc = 0.f, sumK = 0.f;
  for (int q = 0; q < P_N; ++q) {
    float kq = Km[p * 32 + q];
    sumK += kq;
    acc += kq * (grads[q * 8192 + e] - th * z[q * 8192 + e]);
  }
  float zp = z[idx];
  z_new[idx] = zp + (STEP_C / (float)P_N) * (acc + th * zp * sumK);
}

// g_hard = (u_new v_new^T > 0) off-diag
__global__ void k_ghard(const float* __restrict__ znew, float* __restrict__ ghard) {
  int b = blockIdx.x;
  int p = b >> 4, t = b & 15;
  int m0 = (t >> 2) << 4, n0 = (t & 3) << 4;
  const float* zp = znew + p * 8192;
  v8f acc = {};
  acc = wmma_tile(zp, 128, 2, zp + 1, 2, 128, m0, n0, K_N, acc);
  int lane = threadIdx.x & 31;
  int nc = n0 + (lane & 15), mb = (lane >> 4) << 3;
#pragma unroll
  for (int r = 0; r < 8; ++r) {
    int mr = m0 + mb + r;
    ghard[p * 4096 + mr * 64 + nc] = (acc[r] > 0.f && mr != nc) ? 1.f : 0.f;
  }
}

// generic dense layer: C = act(A[MxK] @ W[KxN] + bias)
__global__ void k_mlp(const float* __restrict__ A, const float* __restrict__ W,
                      const float* __restrict__ bias, float* __restrict__ C,
                      int M, int N, int K, int relu) {
  int idx = blockIdx.x * blockDim.x + threadIdx.x;
  if (idx >= M * N) return;
  int m = idx / N, n = idx - m * N;
  float acc = bias[n];
  for (int k = 0; k < K; ++k) acc = fmaf(A[m * K + k], W[k * N + n], acc);
  if (relu) acc = fmaxf(acc, 0.f);
  C[idx] = acc;
}

__global__ void k_qz(const float* __restrict__ qmu, const float* __restrict__ qlv,
                     const int* __restrict__ seedp, float* __restrict__ qz) {
  int idx = blockIdx.x * blockDim.x + threadIdx.x;
  if (idx >= P_N * D_N) return;
  unsigned seed = (unsigned)seedp[0];
  unsigned base = (unsigned)idx * 2u + 0x51EDC6u + seed * 0x9E3779B9u;
  float u1 = fmaxf(u01(pcg_hash(base)), 1e-7f);
  float u2 = u01(pcg_hash(base + 1u));
  float eps = sqrtf(-2.f * logf(u1)) * cosf(6.28318530718f * u2);
  qz[idx] = qmu[idx] + eps * expf(0.5f * qlv[idx]);
}

// ---------------- host-side launch sequence ----------------
extern "C" void kernel_launch(void* const* d_in, const int* in_sizes, int n_in,
                              void* d_out, int out_size, void* d_ws, size_t ws_size,
                              hipStream_t stream) {
  (void)in_sizes; (void)n_in; (void)out_size; (void)ws_size;
  const float* z_gt  = (const float*)d_in[0];
  const float* z0    = (const float*)d_in[1];
  const float* enc_w0 = (const float*)d_in[2];  const float* enc_b0 = (const float*)d_in[3];
  const float* enc_w1 = (const float*)d_in[4];  const float* enc_b1 = (const float*)d_in[5];
  const float* enc_w2 = (const float*)d_in[6];  const float* enc_b2 = (const float*)d_in[7];
  const float* mu_w0  = (const float*)d_in[8];  const float* mu_b0  = (const float*)d_in[9];
  const float* mu_w1  = (const float*)d_in[10]; const float* mu_b1  = (const float*)d_in[11];
  const float* lv_w0  = (const float*)d_in[12]; const float* lv_b0  = (const float*)d_in[13];
  const float* lv_w1  = (const float*)d_in[14]; const float* lv_b1  = (const float*)d_in[15];
  const float* dec_w0 = (const float*)d_in[16]; const float* dec_b0 = (const float*)d_in[17];
  const float* dec_w1 = (const float*)d_in[18]; const float* dec_b1 = (const float*)d_in[19];
  const float* dec_w2 = (const float*)d_in[20]; const float* dec_b2 = (const float*)d_in[21];
  const int*   seedp  = (const int*)d_in[22];

  float* out = (float*)d_out;
  float* ws  = (float*)d_ws;

  // workspace layout (floats)
  float* xbar   = ws + 0;       // 64
  float* R      = ws + 64;      // 4096
  float* scores = ws + 4160;    // 131072
  float* probs  = ws + 135232;  // 131072
  float* logp   = ws + 266304;  // 512
  float* wgt    = ws + 266816;  // 512
  float* btot   = ws + 267328;  // 131072
  float* grads  = ws + 398400;  // 262144
  float* gramz  = ws + 660544;  // 1024
  float* kmat   = ws + 661568;  // 1024
  float* h0     = ws + 662592;  // 640
  float* h1     = ws + 663232;  // 2048
  float* h2     = ws + 665280;  // 2048
  float* tmpb   = ws + 667328;  // 2048
  float* hd0    = ws + 669376;  // 320
  float* hd1    = ws + 669696;  // 4096

  // output layout (flat concat in return order)
  float* o_recons = out + 0;       // 32*128
  float* o_logp   = out + 4096;    // 32
  float* o_qz     = out + 4128;    // 32*64
  float* o_qmu    = out + 6176;    // 32*64
  float* o_qlv    = out + 8224;    // 32*64
  float* o_ghard  = out + 10272;   // 32*64*64
  float* o_znew   = out + 141344;  // 32*64*64*2

  k_xbar      <<<1, 64, 0, stream>>>(z_gt, xbar);
  k_gram_data <<<1, 512, 0, stream>>>(z_gt, R);
  k_finalize_R<<<16, 256, 0, stream>>>(R, xbar);
  k_scores    <<<P_N * 16, 32, 0, stream>>>(z0, scores);
  k_probs     <<<(P_N * 4096 + 255) / 256, 256, 0, stream>>>(scores, probs);
  k_bge       <<<P_N * S_N, 64, 0, stream>>>(R, nullptr, probs, seedp, logp);
  k_softmax   <<<P_N, 32, 0, stream>>>(logp, wgt);
  k_btot      <<<(P_N * 4096 + 255) / 256, 256, 0, stream>>>(probs, wgt, seedp, btot);
  k_grads     <<<P_N * 32, 32, 0, stream>>>(z0, btot, grads);
  k_gram_z    <<<1, 128, 0, stream>>>(z0, gramz);
  k_kmat      <<<4, 256, 0, stream>>>(gramz, kmat);
  k_znew      <<<(P_N * 8192 + 255) / 256, 256, 0, stream>>>(z0, grads, kmat, o_znew);
  k_ghard     <<<P_N * 16, 32, 0, stream>>>(o_znew, o_ghard);
  k_bge       <<<P_N, 64, 0, stream>>>(R, o_ghard, nullptr, seedp, o_logp);

  k_mlp<<<(32 * 20 + 255) / 256, 256, 0, stream>>>(o_ghard, enc_w0, enc_b0, h0, 32, 20, 4096, 1);
  k_mlp<<<8,  256, 0, stream>>>(h0,   enc_w1, enc_b1, h1,     32, 64, 20,  1);
  k_mlp<<<8,  256, 0, stream>>>(h1,   enc_w2, enc_b2, h2,     32, 64, 64,  1);
  k_mlp<<<8,  256, 0, stream>>>(h2,   mu_w0,  mu_b0,  tmpb,   32, 64, 64,  1);
  k_mlp<<<8,  256, 0, stream>>>(tmpb, mu_w1,  mu_b1,  o_qmu,  32, 64, 64,  0);
  k_mlp<<<8,  256, 0, stream>>>(h2,   lv_w0,  lv_b0,  tmpb,   32, 64, 64,  1);
  k_mlp<<<8,  256, 0, stream>>>(tmpb, lv_w1,  lv_b1,  o_qlv,  32, 64, 64,  0);
  k_qz <<<8,  256, 0, stream>>>(o_qmu, o_qlv, seedp, o_qz);
  k_mlp<<<2,  256, 0, stream>>>(o_qz, dec_w0, dec_b0, hd0,      32, 10,  64,  1);
  k_mlp<<<16, 256, 0, stream>>>(hd0,  dec_w1, dec_b1, hd1,      32, 128, 10,  1);
  k_mlp<<<16, 256, 0, stream>>>(hd1,  dec_w2, dec_b2, o_recons, 32, 128, 128, 0);
}
